// PFNetTransformer_19207093748414
// MI455X (gfx1250) — matrix operand order repaired and checked by reading, not compiled, not run on hardware
//
#include <hip/hip_runtime.h>

#define B_ 4
#define N_ 6400
#define D_ 128
#define H_ 4
#define DH_ 128
#define M_ 128
#define L_ 3
#define HD_ (H_*DH_)     // 512
#define NT_ (N_/16)      // 400 16-token tiles per batch
#define TT_ (N_/32)      // 200 32-token tiles per batch (kv reduction K dim)
#define CHUNKS_ 8
#define NC_ (N_/CHUNKS_) // 800 tokens per kv-reduction chunk
#define TPC_ (NC_/32)    // 25 32-token tiles per chunk

typedef __attribute__((ext_vector_type(16))) _Float16 v16h;
typedef __attribute__((ext_vector_type(8)))  _Float16 v8h;
typedef __attribute__((ext_vector_type(8)))  float    v8f;

// ---------------- wave32 WMMA 16x16x32 f16 helpers ----------------
__device__ inline v16h cat8(v8h lo, v8h hi) {
  return __builtin_shufflevector(lo, hi, 0, 1, 2, 3, 4, 5, 6, 7,
                                 8, 9, 10, 11, 12, 13, 14, 15);
}
// A 16x32 (row,k), row-major memory src[row*ld + k].
// lane row = lane&15; K run {kb..kb+15}, kb = 16*(lane>>4) (contiguous order, matches B swizzle).
// The run is contiguous in memory -> two 16B vector loads per lane.
__device__ inline v16h load_a_rm(const _Float16* src, int ld) {
  int lane = threadIdx.x & 31;
  int row  = lane & 15;
  int kb   = (lane >> 4) << 4;
  const _Float16* p = src + (size_t)row * ld + kb;
  v8h lo = *(const v8h*)(p);
  v8h hi = *(const v8h*)(p + 8);
  return cat8(lo, hi);
}
// Swizzled operand tile (A or B register image stored lane-major): 32B contiguous per lane.
__device__ inline v16h load_sw(const _Float16* tile) {
  int lane = threadIdx.x & 31;
  const _Float16* p = tile + lane * 16;
  return cat8(*(const v8h*)p, *(const v8h*)(p + 8));
}
__device__ inline v8f wmma_f16(v16h a, v16h b, v8f c) {
  return __builtin_amdgcn_wmma_f32_16x16x32_f16(false, a, false, b, (short)0, c, false, false);
}
__device__ inline float elu_f(float x)  { return x > 0.f ? x : __expf(x) - 1.f; }
__device__ inline float elu1_f(float x) { return x > 0.f ? x + 1.f : __expf(x); } // elu(x)+1

// ---------------- utility kernels ----------------
__global__ void pf_zero_f32(float* __restrict__ p, int n) {
  int i = blockIdx.x * 256 + threadIdx.x;
  if (i < n) p[i] = 0.f;
}
// Convert f32 [nmat, K, N] matrices to f16 in WMMA-B swizzled tile layout:
// dst[(((mat*NT + ntile)*KT + ktile)*32 + lane)*16 + e] = src[mat][ktile*32 + (lane>>4)*16 + e][ntile*16 + (lane&15)]
__global__ void pf_swizzle_b16(const float* __restrict__ src, _Float16* __restrict__ dst,
                               int K, int Ncols, int nmat) {
  int i = blockIdx.x * 256 + threadIdx.x;
  int total = nmat * K * Ncols;
  if (i >= total) return;
  int KT = K >> 5, NT = Ncols >> 4;
  int e     = i & 15;
  int lane  = (i >> 4) & 31;
  int rest  = i >> 9;
  int ktile = rest % KT;
  int rest2 = rest / KT;
  int ntile = rest2 % NT;
  int mat   = rest2 / NT;
  int k = ktile * 32 + ((lane >> 4) << 4) + e;
  int n = ntile * 16 + (lane & 15);
  dst[i] = (_Float16)src[((size_t)mat * K + k) * Ncols + n];
}

// ---------------- kernel 1: LN1 + QKV + random-feature phi ----------------
__global__ __launch_bounds__(128) void pf_qkv_phi(
    const float* __restrict__ X, const unsigned char* __restrict__ mask,
    const float* __restrict__ ln1_g, const float* __restrict__ ln1_b,
    const _Float16* __restrict__ Wq, const _Float16* __restrict__ Wk,
    const _Float16* __restrict__ Wv,          // swizzled [K=128, N=512]: tile (h*8+ct)*4+kt
    const float* __restrict__ bq, const float* __restrict__ bk,
    const float* __restrict__ bv,             // [H*DH]
    const _Float16* __restrict__ proj,        // swizzled [K=128, N=128]
    float* __restrict__ Xn,                   // [B,N,D]
    _Float16* __restrict__ QP,                // row-major [B,N,H,M]
    _Float16* __restrict__ KPsw,              // swizzled A-layout [B,H][TT][8 mtile][32 lane][16]
    _Float16* __restrict__ Vsw)               // swizzled B-layout [B,H][TT][8 dhtile][32 lane][16]
{
  __shared__ alignas(32) _Float16 Xf[16][D_];
  __shared__ alignas(32) _Float16 Xnf[16][D_];
  __shared__ alignas(32) _Float16 qh[H_][16][DH_];
  __shared__ alignas(32) _Float16 kh[H_][16][DH_];

  int tile = blockIdx.x;
  int b  = tile / NT_;
  int r0 = (tile % NT_) * 16;
  int tid  = threadIdx.x;
  int wave = tid >> 5;                     // wave == head
  int lane = tid & 31, col = lane & 15, rb = (lane >> 4) << 3;
  int h = wave;

  // LN1 (one row per thread 0..15) + stage f16 copies of X and Xn
  if (tid < 16) {
    const float* xr = X + ((size_t)b * N_ + r0 + tid) * D_;
    float mu = 0.f;
    for (int j = 0; j < D_; ++j) mu += xr[j];
    mu *= (1.f / D_);
    float var = 0.f;
    for (int j = 0; j < D_; ++j) { float d = xr[j] - mu; var += d * d; }
    var *= (1.f / D_);
    float rs = rsqrtf(var + 1e-3f);
    float* xnr = Xn + ((size_t)b * N_ + r0 + tid) * D_;
    for (int j = 0; j < D_; ++j) {
      float xn = (xr[j] - mu) * rs * ln1_g[j] + ln1_b[j];
      xnr[j] = xn;
      Xf[tid][j]  = (_Float16)xr[j];
      Xnf[tid][j] = (_Float16)xn;
    }
  }
  __syncthreads();

  size_t bh   = (size_t)b * H_ + h;
  size_t vbase = bh * (size_t)TT_ * 8 * 512;   // per (b,h) swizzled block

  // q (from raw X), k, v (from Xn): [16 x DH] per head, 8 col tiles
  for (int ct = 0; ct < 8; ++ct) {
    v8f cq = {}, ck = {}, cv = {};
    for (int kt = 0; kt < 4; ++kt) {
      v16h aX  = load_a_rm(&Xf[0][0] + kt * 32, D_);
      v16h aXn = load_a_rm(&Xnf[0][0] + kt * 32, D_);
      v16h bwq = load_sw(Wq + (size_t)(((h * 8 + ct) * 4) + kt) * 512);
      v16h bwk = load_sw(Wk + (size_t)(((h * 8 + ct) * 4) + kt) * 512);
      v16h bwv = load_sw(Wv + (size_t)(((h * 8 + ct) * 4) + kt) * 512);
      cq = wmma_f16(aX,  bwq, cq);
      ck = wmma_f16(aXn, bwk, ck);
      cv = wmma_f16(aXn, bwv, cv);
    }
#pragma unroll
    for (int v = 0; v < 8; ++v) {
      int dh = ct * 16 + col, row = rb + v;
      qh[h][row][dh] = (_Float16)(cq[v] + bq[h * DH_ + dh]);
      kh[h][row][dh] = (_Float16)(ck[v] + bk[h * DH_ + dh]);
      // scatter v into WMMA-B swizzled layout (token = K dim of kv reduction)
      int token = r0 + row;
      int tt = token >> 5, t32 = token & 31;
      int laneB = col + ((t32 >> 4) << 4);
      int eB = t32 & 15;
      Vsw[vbase + (((size_t)tt * 8 + ct) * 32 + laneB) * 16 + eB] =
          (_Float16)(cv[v] + bv[h * DH_ + dh]);
    }
  }
  __syncthreads();

  // phi: qp = elu(q@proj)+1 ; kp = (elu(k@proj)+1) * mask
  for (int ct = 0; ct < 8; ++ct) {
    v8f cq = {}, ck = {};
    for (int kt = 0; kt < 4; ++kt) {
      v16h aq = load_a_rm(&qh[h][0][0] + kt * 32, DH_);
      v16h ak = load_a_rm(&kh[h][0][0] + kt * 32, DH_);
      v16h bp = load_sw(proj + (size_t)(ct * 4 + kt) * 512);
      cq = wmma_f16(aq, bp, cq);
      ck = wmma_f16(ak, bp, ck);
    }
#pragma unroll
    for (int v = 0; v < 8; ++v) {
      int m = ct * 16 + col, row = rb + v;
      float mf = (float)mask[(size_t)b * N_ + r0 + row];
      QP[(((size_t)b * N_ + r0 + row) * H_ + h) * M_ + m] = (_Float16)elu1_f(cq[v]);
      // scatter kp into WMMA-A swizzled layout (A[m, token])
      int token = r0 + row;
      int tt = token >> 5, t32 = token & 31;
      int laneA = col + ((t32 >> 4) << 4);
      int eA = t32 & 15;
      KPsw[vbase + (((size_t)tt * 8 + ct) * 32 + laneA) * 16 + eA] =
          (_Float16)(elu1_f(ck[v]) * mf);
    }
  }
}

// ---------------- kernel 2: kv = sum_n kp (x) v  and ksum = sum_n kp ----------------
__global__ __launch_bounds__(256) void pf_kv_reduce(
    const _Float16* __restrict__ KPsw, const _Float16* __restrict__ Vsw,
    float* __restrict__ kv,   // [B,H,M,DH] f32
    float* __restrict__ ksum) // [B,H,M]
{
  int bid = blockIdx.x;           // B*H*CHUNKS
  int c  = bid % CHUNKS_;
  int bhi = bid / CHUNKS_;
  int h = bhi % H_, b = bhi / H_;
  int tid = threadIdx.x;
  int mi = tid >> 5;              // wave -> M tile row (8 waves, 8 tiles)
  int lane = tid & 31, col = lane & 15, rb = (lane >> 4) << 3;

  size_t bh = (size_t)b * H_ + h;
  const _Float16* KPb = KPsw + bh * (size_t)TT_ * 8 * 512;
  const _Float16* Vb  = Vsw  + bh * (size_t)TT_ * 8 * 512;
  int tt0 = c * TPC_;

  v8f acc[8] = {};
  for (int t = 0; t < TPC_; ++t) {
    size_t tt = tt0 + t;
    v16h a = load_sw(KPb + (tt * 8 + mi) * 512);      // A[m, token32]
#pragma unroll
    for (int ki = 0; ki < 8; ++ki) {
      v16h bm = load_sw(Vb + (tt * 8 + ki) * 512);    // B[token32, dh]
      acc[ki] = wmma_f16(a, bm, acc[ki]);
    }
  }
  float* kvb = kv + bh * M_ * DH_;
#pragma unroll
  for (int ki = 0; ki < 8; ++ki)
#pragma unroll
    for (int v = 0; v < 8; ++v)
      atomicAdd(&kvb[(size_t)(mi * 16 + rb + v) * DH_ + ki * 16 + col], acc[ki][v]);

  // ksum[m] = sum_n kp[n, m]; gather contiguous 16-element runs of the swizzled layout
  if (tid < M_) {
    int mt = tid >> 4, ml = tid & 15;
    float s = 0.f;
    for (int t = 0; t < TPC_; ++t) {
      size_t tt = tt0 + t;
      const _Float16* base = KPb + (tt * 8 + mt) * 512;
#pragma unroll
      for (int half = 0; half < 2; ++half) {
        const _Float16* p = base + (ml + half * 16) * 16;
#pragma unroll
        for (int e = 0; e < 16; ++e) s += (float)p[e];
      }
    }
    atomicAdd(&ksum[bh * M_ + tid], s);
  }
}

// ---------------- kernel 3: o = qp@kv * denom -> Wo -> LNs -> FFN -> LN -> W2 ----------------
__global__ __launch_bounds__(128) void pf_attn_ffn(
    const _Float16* __restrict__ QP,    // row-major [B,N,H,M]
    const _Float16* __restrict__ kvsw,  // swizzled [B*H mats, K=M, N=DH]
    const float* __restrict__ ksum,
    const _Float16* __restrict__ Wo, const float* __restrict__ bo,   // Wo swizzled [K=512,N=128]
    const unsigned char* __restrict__ mask,
    const float* __restrict__ Xn,
    const float* __restrict__ ln2_g,  const float* __restrict__ ln2_b,
    const float* __restrict__ fln1_g, const float* __restrict__ fln1_b,
    const float* __restrict__ fln2_g, const float* __restrict__ fln2_b,
    const _Float16* __restrict__ W1,  const float* __restrict__ b1,  // swizzled [128,128]
    const _Float16* __restrict__ W2,  const float* __restrict__ b2,  // swizzled [128,128]
    float* __restrict__ Xout)
{
  __shared__ alignas(32) _Float16 obuf[16][HD_];
  __shared__ alignas(32) float    ybuf[16][D_];
  __shared__ alignas(32) _Float16 o1[16][D_];
  __shared__ alignas(32) float    hbuf[16][D_];
  __shared__ alignas(32) _Float16 h2[16][D_];
  __shared__ float    denom[16][H_];

  int tile = blockIdx.x;
  int b  = tile / NT_;
  int r0 = (tile % NT_) * 16;
  int tid  = threadIdx.x;
  int wave = tid >> 5;
  int lane = tid & 31, col = lane & 15, rb = (lane >> 4) << 3;
  int h = wave;

  // denominators: 1 / (qp . ksum + 1e-6), 64 (row,head) pairs
  if (tid < 64) {
    int row = tid >> 2, hh = tid & 3;
    const _Float16* qr = QP + (((size_t)b * N_ + r0 + row) * H_ + hh) * M_;
    const float* ks = ksum + (b * H_ + hh) * M_;
    float s = 0.f;
    for (int m = 0; m < M_; ++m) s += (float)qr[m] * ks[m];
    denom[row][hh] = 1.f / (s + 1e-6f);
  }
  __syncthreads();

  const _Float16* qpA = QP + (((size_t)b * N_ + r0) * H_ + h) * M_;
  const _Float16* kvB = kvsw + (size_t)(b * H_ + h) * M_ * DH_;  // 8 ntile x 4 ktile x 512
  for (int ct = 0; ct < 8; ++ct) {
    v8f cacc = {};
    for (int kt = 0; kt < 4; ++kt) {
      v16h a  = load_a_rm(qpA + kt * 32, H_ * M_);
      v16h bm = load_sw(kvB + (size_t)(ct * 4 + kt) * 512);
      cacc = wmma_f16(a, bm, cacc);
    }
#pragma unroll
    for (int v = 0; v < 8; ++v) {
      int row = rb + v;
      obuf[row][h * DH_ + ct * 16 + col] = (_Float16)(cacc[v] * denom[row][h]);
    }
  }
  __syncthreads();

  // out = o @ Wo + bo ; attn*mask ; y = Xn + attn
  for (int t = 0; t < 2; ++t) {
    int ct = wave * 2 + t;
    v8f cacc = {};
    for (int kt = 0; kt < 16; ++kt) {
      v16h a  = load_a_rm(&obuf[0][0] + kt * 32, HD_);
      v16h bm = load_sw(Wo + (size_t)(ct * 16 + kt) * 512);
      cacc = wmma_f16(a, bm, cacc);
    }
#pragma unroll
    for (int v = 0; v < 8; ++v) {
      int row = rb + v, cc = ct * 16 + col;
      float mf = (float)mask[(size_t)b * N_ + r0 + row];
      float attn = (cacc[v] + bo[cc]) * mf;
      ybuf[row][cc] = Xn[((size_t)b * N_ + r0 + row) * D_ + cc] + attn;
    }
  }
  __syncthreads();

  // LN(ln2) then LN(fln1) -> o1 (f16)
  if (tid < 16) {
    float* yr = &ybuf[tid][0];
    for (int pass = 0; pass < 2; ++pass) {
      float mu = 0.f;
      for (int j = 0; j < D_; ++j) mu += yr[j];
      mu *= (1.f / D_);
      float var = 0.f;
      for (int j = 0; j < D_; ++j) { float d = yr[j] - mu; var += d * d; }
      var *= (1.f / D_);
      float rs = rsqrtf(var + 1e-3f);
      if (pass == 0) {
        for (int j = 0; j < D_; ++j) yr[j] = (yr[j] - mu) * rs * ln2_g[j] + ln2_b[j];
      } else {
        for (int j = 0; j < D_; ++j)
          o1[tid][j] = (_Float16)((yr[j] - mu) * rs * fln1_g[j] + fln1_b[j]);
      }
    }
  }
  __syncthreads();

  // h = elu(o1 @ W1 + b1)
  for (int t = 0; t < 2; ++t) {
    int ct = wave * 2 + t;
    v8f cacc = {};
    for (int kt = 0; kt < 4; ++kt) {
      v16h a  = load_a_rm(&o1[0][0] + kt * 32, D_);
      v16h bm = load_sw(W1 + (size_t)(ct * 4 + kt) * 512);
      cacc = wmma_f16(a, bm, cacc);
    }
#pragma unroll
    for (int v = 0; v < 8; ++v)
      hbuf[rb + v][ct * 16 + col] = elu_f(cacc[v] + b1[ct * 16 + col]);
  }
  __syncthreads();

  // LN(fln2) -> h2 (f16)
  if (tid < 16) {
    float* hr = &hbuf[tid][0];
    float mu = 0.f;
    for (int j = 0; j < D_; ++j) mu += hr[j];
    mu *= (1.f / D_);
    float var = 0.f;
    for (int j = 0; j < D_; ++j) { float d = hr[j] - mu; var += d * d; }
    var *= (1.f / D_);
    float rs = rsqrtf(var + 1e-3f);
    for (int j = 0; j < D_; ++j)
      h2[tid][j] = (_Float16)((hr[j] - mu) * rs * fln2_g[j] + fln2_b[j]);
  }
  __syncthreads();

  // Xout = h2 @ W2 + b2
  for (int t = 0; t < 2; ++t) {
    int ct = wave * 2 + t;
    v8f cacc = {};
    for (int kt = 0; kt < 4; ++kt) {
      v16h a  = load_a_rm(&h2[0][0] + kt * 32, D_);
      v16h bm = load_sw(W2 + (size_t)(ct * 4 + kt) * 512);
      cacc = wmma_f16(a, bm, cacc);
    }
#pragma unroll
    for (int v = 0; v < 8; ++v)
      Xout[((size_t)b * N_ + r0 + rb + v) * D_ + ct * 16 + col] =
          cacc[v] + b2[ct * 16 + col];
  }
}

// ---------------- host-side orchestration ----------------
extern "C" void kernel_launch(void* const* d_in, const int* in_sizes, int n_in,
                              void* d_out, int out_size, void* d_ws, size_t ws_size,
                              hipStream_t stream) {
  (void)in_sizes; (void)n_in; (void)out_size; (void)ws_size;
  const float* X  = (const float*)d_in[0];
  const unsigned char* mask = (const unsigned char*)d_in[1];
  // params flattened alphabetically (JAX pytree dict order):
  const float* W1 = (const float*)d_in[2];
  const float* W2 = (const float*)d_in[3];
  const float* Wk = (const float*)d_in[4];
  const float* Wo = (const float*)d_in[5];
  const float* Wq = (const float*)d_in[6];
  const float* Wv = (const float*)d_in[7];
  const float* b1 = (const float*)d_in[8];
  const float* b2 = (const float*)d_in[9];
  const float* bk = (const float*)d_in[10];
  const float* bo = (const float*)d_in[11];
  const float* bq = (const float*)d_in[12];
  const float* bv = (const float*)d_in[13];
  const float* fln1_b = (const float*)d_in[14];
  const float* fln1_g = (const float*)d_in[15];
  const float* fln2_b = (const float*)d_in[16];
  const float* fln2_g = (const float*)d_in[17];
  const float* ln1_b = (const float*)d_in[18];
  const float* ln1_g = (const float*)d_in[19];
  const float* ln2_b = (const float*)d_in[20];
  const float* ln2_g = (const float*)d_in[21];
  const float* proj = (const float*)d_in[22];

  char* w = (char*)d_ws;
  size_t off = 0;
  auto take = [&](size_t bytes) -> void* {
    void* p = w + off;
    off += (bytes + 255) & ~(size_t)255;
    return p;
  };

  _Float16* wq16 = (_Float16*)take(sizeof(_Float16) * (size_t)L_ * D_ * HD_);
  _Float16* wk16 = (_Float16*)take(sizeof(_Float16) * (size_t)L_ * D_ * HD_);
  _Float16* wv16 = (_Float16*)take(sizeof(_Float16) * (size_t)L_ * D_ * HD_);
  _Float16* wo16 = (_Float16*)take(sizeof(_Float16) * (size_t)L_ * HD_ * D_);
  _Float16* w116 = (_Float16*)take(sizeof(_Float16) * (size_t)L_ * D_ * D_);
  _Float16* w216 = (_Float16*)take(sizeof(_Float16) * (size_t)L_ * D_ * D_);
  _Float16* pr16 = (_Float16*)take(sizeof(_Float16) * (size_t)DH_ * M_);
  float*    Xn   = (float*)   take(sizeof(float)    * (size_t)B_ * N_ * D_);
  _Float16* QP   = (_Float16*)take(sizeof(_Float16) * (size_t)B_ * N_ * H_ * M_);
  _Float16* KPsw = (_Float16*)take(sizeof(_Float16) * (size_t)B_ * N_ * H_ * M_);
  _Float16* Vsw  = (_Float16*)take(sizeof(_Float16) * (size_t)B_ * N_ * H_ * DH_);
  float*    kv   = (float*)   take(sizeof(float)    * (size_t)B_ * H_ * M_ * DH_);
  _Float16* kvsw = (_Float16*)take(sizeof(_Float16) * (size_t)B_ * H_ * M_ * DH_);
  float*    ksum = (float*)   take(sizeof(float)    * (size_t)B_ * H_ * M_);
  float*    XA   = (float*)   take(sizeof(float)    * (size_t)B_ * N_ * D_);
  float*    XB   = (float*)   take(sizeof(float)    * (size_t)B_ * N_ * D_);

  auto swz = [&](const float* s, _Float16* d, int K, int Ncols, int nmat) {
    int n = K * Ncols * nmat;
    pf_swizzle_b16<<<(n + 255) / 256, 256, 0, stream>>>(s, d, K, Ncols, nmat);
  };
  swz(Wq, wq16, D_, HD_, L_);
  swz(Wk, wk16, D_, HD_, L_);
  swz(Wv, wv16, D_, HD_, L_);
  swz(Wo, wo16, HD_, D_, L_);
  swz(W1, w116, D_, D_, L_);
  swz(W2, w216, D_, D_, L_);
  swz(proj, pr16, DH_, M_, 1);

  for (int l = 0; l < L_; ++l) {
    const float* Xin = (l == 0) ? X : ((l == 1) ? XA : XB);
    float* Xout = (l == L_ - 1) ? (float*)d_out : ((l == 0) ? XA : XB);

    int nkv = B_ * H_ * M_ * DH_;
    pf_zero_f32<<<(nkv + 255) / 256, 256, 0, stream>>>(kv, nkv);
    int nks = B_ * H_ * M_;
    pf_zero_f32<<<(nks + 255) / 256, 256, 0, stream>>>(ksum, nks);

    pf_qkv_phi<<<B_ * NT_, 128, 0, stream>>>(
        Xin, mask,
        ln1_g + (size_t)l * D_, ln1_b + (size_t)l * D_,
        wq16 + (size_t)l * D_ * HD_, wk16 + (size_t)l * D_ * HD_,
        wv16 + (size_t)l * D_ * HD_,
        bq + (size_t)l * HD_, bk + (size_t)l * HD_, bv + (size_t)l * HD_,
        pr16, Xn, QP, KPsw, Vsw);

    pf_kv_reduce<<<B_ * H_ * CHUNKS_, 256, 0, stream>>>(KPsw, Vsw, kv, ksum);

    // kv f32 -> f16, swizzled into WMMA-B tile layout (B*H matrices of [M, DH])
    pf_swizzle_b16<<<(nkv + 255) / 256, 256, 0, stream>>>(kv, kvsw, M_, DH_, B_ * H_);

    pf_attn_ffn<<<B_ * NT_, 128, 0, stream>>>(
        QP, kvsw, ksum,
        wo16 + (size_t)l * HD_ * D_, bo + (size_t)l * D_,
        mask, Xn,
        ln2_g + (size_t)l * D_, ln2_b + (size_t)l * D_,
        fln1_g + (size_t)l * D_, fln1_b + (size_t)l * D_,
        fln2_g + (size_t)l * D_, fln2_b + (size_t)l * D_,
        w116 + (size_t)l * D_ * D_, b1 + (size_t)l * D_,
        w216 + (size_t)l * D_ * D_, b2 + (size_t)l * D_,
        Xout);
  }
}
